// RwkvEncoder_70935679860890
// MI455X (gfx1250) — matrix operand order repaired and checked by reading, not compiled, not run on hardware
//
#include <hip/hip_runtime.h>

// ---------------------------------------------------------------------------
// RWKV-6 style layer for gfx1250 (MI455X).
//  - Dense projections: v_wmma_f32_16x16x32_bf16, B-operand staged to LDS by
//    the Tensor Data Mover (tensor_load_to_lds, double buffered, TENSORcnt).
//  - Chunked recurrent scan: v_wmma_f32_16x16x4_f32 (full fp32), state in LDS,
//    dual accumulator chains to overlap ds_load latency with matrix ops.
// B=2 L=2048 H=1024 NH=4 K=512 V=1024 DK=128 DV=256 PR=32 GR=64
// ---------------------------------------------------------------------------

typedef __attribute__((ext_vector_type(16))) __bf16       v16bf;
typedef __attribute__((ext_vector_type(8)))  float        v8f;
typedef __attribute__((ext_vector_type(2)))  float        v2f;
typedef __attribute__((ext_vector_type(4)))  unsigned int u32x4;
typedef __attribute__((ext_vector_type(4)))  unsigned int v4u;
typedef __attribute__((ext_vector_type(8)))  int          v8i;
typedef __attribute__((ext_vector_type(4)))  int          v4i;

union BFfrag { v16bf v; u32x4 q[2]; };

#if __has_builtin(__builtin_amdgcn_tensor_load_to_lds) && \
    __has_builtin(__builtin_amdgcn_s_wait_tensorcnt)
#define HAVE_TDM 1
#else
#define HAVE_TDM 0
#endif

__device__ __forceinline__ unsigned short f2bf(float f) {
  unsigned int u = __builtin_bit_cast(unsigned int, f);
  u += 0x7fffu + ((u >> 16) & 1u);            // round-to-nearest-even
  return (unsigned short)(u >> 16);
}

#define BLr   4096          // B*L rows
#define Hd    1024
#define Kd    512
#define Vd    1024
#define DKs   128
#define DVs   256
#define CH    32            // scan chunk length
#define NCHNK 64            // L / CH

// ---------------------------------------------------------------------------
// fp32 -> bf16 conversion (weights, once per call)
// ---------------------------------------------------------------------------
__global__ void f32_to_bf16_kernel(const float* __restrict__ src,
                                   unsigned short* __restrict__ dst, int n) {
  int i = blockIdx.x * blockDim.x + threadIdx.x;
  if (i < n) dst[i] = f2bf(src[i]);
}

// ---------------------------------------------------------------------------
// token shift: delta = shift(x) - x ; t0 = bf16(x + delta*mu_x)
// ---------------------------------------------------------------------------
__global__ void prep_kernel(const float* __restrict__ x,
                            const float* __restrict__ mu_x,
                            float* __restrict__ delta,
                            unsigned short* __restrict__ t0b) {
  int i = blockIdx.x * blockDim.x + threadIdx.x;
  if (i >= BLr * Hd) return;
  int row = i >> 10;
  int t   = row & 2047;             // position within sequence (L=2048)
  int col = i & 1023;
  float xv = x[i];
  float sh = (t == 0) ? 0.f : x[i - Hd];
  float d  = sh - xv;
  delta[i] = d;
  t0b[i]   = f2bf(xv + d * mu_x[col]);
}

// ---------------------------------------------------------------------------
// TDM: async 2-D tile load (tile_dim0=32 elems of 2B along k, tile_dim1=64
// rows) from global weight matrix into LDS, row-major [64][32] halves.
// D# packing per CDNA5 ISA ch.8 (group0 128b, group1 256b, groups 2/3 zero).
// This toolchain exposes the 6-arg builtin (extra zero-filled i32x8 group).
// ---------------------------------------------------------------------------
#if HAVE_TDM
__device__ __forceinline__ void tdm_load_b_tile(unsigned lds_byte,
                                                const unsigned short* gsrc,
                                                unsigned tensor_d0,
                                                unsigned tensor_d1,
                                                unsigned long long stride0) {
  unsigned long long ga = (unsigned long long)gsrc;
  v4u g0;
  g0[0] = 1u;                                   // count=1, user descriptor
  g0[1] = lds_byte;                             // lds_addr
  g0[2] = (unsigned)ga;                         // global_addr[31:0]
  g0[3] = ((unsigned)(ga >> 32) & 0x01ffffffu)  // global_addr[56:32]
          | 0x80000000u;                        // type=2 ("image")
  v8i g1;
  g1[0] = (int)(1u << 16);                      // data_size=1 -> 2 bytes
  g1[1] = (int)((tensor_d0 & 0xffffu) << 16);   // tensor_dim0[15:0]
  g1[2] = (int)((tensor_d0 >> 16) | ((tensor_d1 & 0xffffu) << 16));
  g1[3] = (int)((tensor_d1 >> 16) | (32u << 16)); // tile_dim0 = 32
  g1[4] = (int)64u;                             // tile_dim1 = 64, tile_dim2=0
  g1[5] = (int)(unsigned)(stride0 & 0xffffffffull);
  g1[6] = (int)(unsigned)((stride0 >> 32) & 0xffffull);
  g1[7] = 0;
  v4i gz4 = {0, 0, 0, 0};
  v8i gz8 = {0, 0, 0, 0, 0, 0, 0, 0};
  __builtin_amdgcn_tensor_load_to_lds(g0, g1, gz4, gz4, gz8, 0);
}
#endif

// ---------------------------------------------------------------------------
// Generic WMMA bf16 GEMM:  C = epi(A[M,K] @ W[N,K]^T)
//  EPI 0: f32 out (+bias if non-null)
//  EPI 1: bf16 out = tanh(acc)
//  EPI 2: bf16 out = x + delta*(acc + bias)        (lerp-mix fusion, N==1024)
// Workgroup: 256 thr = 8 waves; tile 128(M) x 64(N); K step 32.
// B slab (32k x 64n) staged to LDS by TDM, double buffered.
// ---------------------------------------------------------------------------
template <int EPI, typename OutT>
__global__ __launch_bounds__(256)
void gemm_wmma(const unsigned short* __restrict__ A, int lda,
               const unsigned short* __restrict__ W, int ldw,
               OutT* __restrict__ C, int ldc,
               int M, int N, int K,
               const float* __restrict__ bias,
               const float* __restrict__ xres,
               const float* __restrict__ dres) {
  const int wave = threadIdx.x >> 5;
  const int lane = threadIdx.x & 31;
  const int lrow = lane & 15;
  const int lhi  = lane >> 4;
  const int m0   = blockIdx.x * 128 + wave * 16;
  const int n0   = blockIdx.y * 64;

  v8f acc[4] = {};

#if HAVE_TDM
  __shared__ unsigned short Bs[2][64 * 32];     // double buffered B slab, 8KB
  const int nsteps = K >> 5;
  if (wave == 0) {
    tdm_load_b_tile((unsigned)(unsigned long long)(void*)&Bs[0][0],
                    W + (size_t)n0 * ldw, (unsigned)K, (unsigned)N,
                    (unsigned long long)ldw);
  }
  for (int s = 0; s < nsteps; ++s) {
    const int k0 = s << 5;
    if (wave == 0) __builtin_amdgcn_s_wait_tensorcnt((short)0);
    __syncthreads();                            // buffer s&1 ready for all
    if (wave == 0 && s + 1 < nsteps) {
      tdm_load_b_tile((unsigned)(unsigned long long)(void*)&Bs[(s + 1) & 1][0],
                      W + (size_t)n0 * ldw + (k0 + 32), (unsigned)K,
                      (unsigned)N, (unsigned long long)ldw);
    }
    // A fragment: row = m0+lrow, halves j<8 -> k = lhi*8+j ; j>=8 -> +16
    BFfrag af;
    const unsigned short* ap = A + (size_t)(m0 + lrow) * lda + k0 + lhi * 8;
    af.q[0] = *(const u32x4*)(ap);
    af.q[1] = *(const u32x4*)(ap + 16);
    const unsigned short* bbase = &Bs[s & 1][0];
#pragma unroll
    for (int t = 0; t < 4; ++t) {
      BFfrag bf;
      const unsigned short* bp = bbase + (t * 16 + lrow) * 32 + lhi * 8;
      bf.q[0] = *(const u32x4*)(bp);
      bf.q[1] = *(const u32x4*)(bp + 16);
      acc[t] = __builtin_amdgcn_wmma_f32_16x16x32_bf16(
          false, af.v, false, bf.v, (short)0, acc[t], false, false);
    }
    __syncthreads();                            // reads done before re-fill
  }
#else
  for (int k0 = 0; k0 < K; k0 += 32) {
    BFfrag af;
    const unsigned short* ap = A + (size_t)(m0 + lrow) * lda + k0 + lhi * 8;
    af.q[0] = *(const u32x4*)(ap);
    af.q[1] = *(const u32x4*)(ap + 16);
#pragma unroll
    for (int t = 0; t < 4; ++t) {
      int n = n0 + t * 16 + lrow;
      if (n > N - 1) n = N - 1;
      BFfrag bf;
      const unsigned short* bp = W + (size_t)n * ldw + k0 + lhi * 8;
      bf.q[0] = *(const u32x4*)(bp);
      bf.q[1] = *(const u32x4*)(bp + 16);
      acc[t] = __builtin_amdgcn_wmma_f32_16x16x32_bf16(
          false, af.v, false, bf.v, (short)0, acc[t], false, false);
    }
  }
#endif

#pragma unroll
  for (int t = 0; t < 4; ++t) {
    int col = n0 + t * 16 + lrow;
    if (col >= N) continue;
#pragma unroll
    for (int i = 0; i < 8; ++i) {
      int row = m0 + lhi * 8 + i;
      float v = acc[t][i];
      if constexpr (EPI == 0) {
        if (bias) v += bias[col];
        C[(size_t)row * ldc + col] = v;
      } else if constexpr (EPI == 1) {
        C[(size_t)row * ldc + col] = (OutT)f2bf(tanhf(v));
      } else {
        v += bias[col];
        size_t ix = (size_t)row * Hd + col;     // x/delta are [BL, H]
        C[(size_t)row * ldc + col] = (OutT)f2bf(xres[ix] + dres[ix] * v);
      }
    }
  }
}

// ---------------------------------------------------------------------------
// Chunked recurrent scan. One workgroup per (batch, head); state S in LDS.
// Full-precision fp32 WMMA (16x16x4), two independent accumulator chains
// per tile so ds_load latency overlaps WMMA issue.
// ---------------------------------------------------------------------------
__device__ __forceinline__ v8f wmma_f32(v2f a, v2f b, v8f c) {
  return __builtin_amdgcn_wmma_f32_16x16x4_f32(false, a, false, b, (short)0, c,
                                               false, false);
}

#define SCAN_SMEM_FLOATS (32768 + 8192 + 4096 * 4 + 1024 + 128 + 128)

__global__ __launch_bounds__(256)
void scan_kernel(const float* __restrict__ rg, const float* __restrict__ kg,
                 const float* __restrict__ wg, const float* __restrict__ vg,
                 const float* __restrict__ bonus, float* __restrict__ og) {
  extern __shared__ float sm[];
  float* S   = sm;                 // [128][256] running state
  float* Vc  = S + 32768;          // [32][256] value chunk
  float* Rt  = Vc + 8192;          // [32][128] r * exp(cw_excl)
  float* Kt  = Rt + 4096;          // [32][128] k * exp(-cw_incl)
  float* Ke  = Kt + 4096;          // [32][128] k * exp(cw_tot - cw_incl)
  float* Cw  = Ke + 4096;          // [32][128] inclusive cumsum of log-decay
  float* Mm  = Cw + 4096;          // [32][32]  intra-chunk attention
  float* Ct  = Mm + 1024;          // [128]     cw_tot per key dim
  float* Dec = Ct + 128;           // [128]     exp(cw_tot)

  const int bh   = blockIdx.x;
  const int b    = bh >> 2, h = bh & 3;
  const int tid  = threadIdx.x;
  const int wave = tid >> 5, lane = tid & 31;
  const int lrow = lane & 15, lhi = lane >> 4;
  const float* uvec = bonus + h * DKs;

  for (int i = tid; i < 32768; i += 256) S[i] = 0.f;
  __syncthreads();

  for (int c = 0; c < NCHNK; ++c) {
    const long rowc = (long)b * 2048 + (long)c * CH;

    // load log-decay w_t = -exp(w_raw) and value chunk
    for (int i = tid; i < CH * DKs; i += 256) {
      int t = i >> 7, d = i & 127;
      Cw[i] = -__expf(wg[(rowc + t) * Kd + h * DKs + d]);
    }
    for (int i = tid; i < CH * DVs; i += 256) {
      int t = i >> 8, dv = i & 255;
      Vc[i] = vg[(rowc + t) * Vd + h * DVs + dv];
    }
    __syncthreads();

    // per-channel cumulative log-decay
    if (tid < DKs) {
      float cum = 0.f;
      for (int t = 0; t < CH; ++t) { cum += Cw[t * DKs + tid]; Cw[t * DKs + tid] = cum; }
      Ct[tid]  = cum;
      Dec[tid] = __expf(cum);
    }
    __syncthreads();

    // scaled r / k variants
    for (int i = tid; i < CH * DKs; i += 256) {
      int t = i >> 7, d = i & 127;
      float cwI = Cw[i];
      float cwE = (t == 0) ? 0.f : Cw[i - DKs];
      float rv = rg[(rowc + t) * Kd + h * DKs + d];
      float kv = kg[(rowc + t) * Kd + h * DKs + d];
      Rt[i] = rv * __expf(cwE);
      Kt[i] = kv * __expf(-cwI);
      Ke[i] = kv * __expf(Ct[d] - cwI);
    }
    __syncthreads();

    // M = strict-lower(Rt Kt^T) + diag(r . u*k)
    if (wave < 4) {
      int ti = wave >> 1, tj = wave & 1;
      const int rA = (ti * 16 + lrow) * DKs;
      const int rB = (tj * 16 + lrow) * DKs;
      v8f a0 = {}, a1 = {};
      for (int kk = 0; kk < DKs; kk += 8) {
        int ka = kk + lhi * 2;
        v2f x0, y0, x1, y1;
        x0.x = Rt[rA + ka];     x0.y = Rt[rA + ka + 1];
        y0.x = Kt[rB + ka];     y0.y = Kt[rB + ka + 1];
        x1.x = Rt[rA + ka + 4]; x1.y = Rt[rA + ka + 5];
        y1.x = Kt[rB + ka + 4]; y1.y = Kt[rB + ka + 5];
        a0 = wmma_f32(x0, y0, a0);
        a1 = wmma_f32(x1, y1, a1);
      }
      v8f acc = a0 + a1;
#pragma unroll
      for (int i = 0; i < 8; ++i) {
        int rr = ti * 16 + lhi * 8 + i, cc = tj * 16 + lrow;
        if (cc != rr) Mm[rr * CH + cc] = (cc < rr) ? acc[i] : 0.f;
      }
    } else if (wave == 4) {
      int t = lane;                               // 32 diagonal elements
      float s = 0.f;
      const float* rp = rg + (rowc + t) * Kd + h * DKs;
      const float* kp = kg + (rowc + t) * Kd + h * DKs;
      for (int d = 0; d < DKs; ++d) s += rp[d] * uvec[d] * kp[d];
      Mm[t * CH + t] = s;
    }
    __syncthreads();

    // O[32][256] = Mm @ Vc + Rt @ S_in   (32 tiles of 16x16, 4 per wave)
    for (int q = 0; q < 4; ++q) {
      int idx = q * 8 + wave;
      int ti = idx & 1, tj = idx >> 1;
      const int rM = (ti * 16 + lrow) * CH;
      const int rR = (ti * 16 + lrow) * DKs;
      const int cc = tj * 16 + lrow;
      v8f a0 = {}, a1 = {};
      for (int kk = 0; kk < CH; kk += 8) {
        int ka = kk + lhi * 2;
        v2f x0, y0, x1, y1;
        x0.x = Mm[rM + ka];     x0.y = Mm[rM + ka + 1];
        y0.x = Vc[ka * DVs + cc];       y0.y = Vc[(ka + 1) * DVs + cc];
        x1.x = Mm[rM + ka + 4]; x1.y = Mm[rM + ka + 5];
        y1.x = Vc[(ka + 4) * DVs + cc]; y1.y = Vc[(ka + 5) * DVs + cc];
        a0 = wmma_f32(x0, y0, a0);
        a1 = wmma_f32(x1, y1, a1);
      }
      for (int kk = 0; kk < DKs; kk += 8) {
        int ka = kk + lhi * 2;
        v2f x0, y0, x1, y1;
        x0.x = Rt[rR + ka];     x0.y = Rt[rR + ka + 1];
        y0.x = S[ka * DVs + cc];        y0.y = S[(ka + 1) * DVs + cc];
        x1.x = Rt[rR + ka + 4]; x1.y = Rt[rR + ka + 5];
        y1.x = S[(ka + 4) * DVs + cc];  y1.y = S[(ka + 5) * DVs + cc];
        a0 = wmma_f32(x0, y0, a0);
        a1 = wmma_f32(x1, y1, a1);
      }
      v8f acc = a0 + a1;
#pragma unroll
      for (int i = 0; i < 8; ++i) {
        long orow = rowc + ti * 16 + lhi * 8 + i;
        og[orow * Vd + h * DVs + cc] = acc[i];
      }
    }
    __syncthreads();

    // S = diag(Dec) S + Ke^T @ Vc   (128 tiles of 16x16, 16 per wave)
    for (int q = 0; q < 16; ++q) {
      int idx = q * 8 + wave;
      int ti = idx & 7, tj = idx >> 3;
      const int cc = tj * 16 + lrow;
      v8f a0, a1 = {};
#pragma unroll
      for (int i = 0; i < 8; ++i) {
        int dd = ti * 16 + lhi * 8 + i;
        a0[i] = S[dd * DVs + cc] * Dec[dd];
      }
      for (int kk = 0; kk < CH; kk += 8) {
        int ka = kk + lhi * 2;
        v2f x0, y0, x1, y1;
        x0.x = Ke[ka * DKs + ti * 16 + lrow];
        x0.y = Ke[(ka + 1) * DKs + ti * 16 + lrow];
        y0.x = Vc[ka * DVs + cc];       y0.y = Vc[(ka + 1) * DVs + cc];
        x1.x = Ke[(ka + 4) * DKs + ti * 16 + lrow];
        x1.y = Ke[(ka + 5) * DKs + ti * 16 + lrow];
        y1.x = Vc[(ka + 4) * DVs + cc]; y1.y = Vc[(ka + 5) * DVs + cc];
        a0 = wmma_f32(x0, y0, a0);
        a1 = wmma_f32(x1, y1, a1);
      }
      v8f acc = a0 + a1;
#pragma unroll
      for (int i = 0; i < 8; ++i) {
        int dd = ti * 16 + lhi * 8 + i;
        S[dd * DVs + cc] = acc[i];
      }
    }
    __syncthreads();
  }
}

// ---------------------------------------------------------------------------
// per-head LayerNorm + swish gate -> bf16
// one wave per (row, head); 256 elems, 8 per lane, shuffle reduce
// ---------------------------------------------------------------------------
__global__ __launch_bounds__(256)
void lngate_kernel(const float* __restrict__ o, const float* __restrict__ g,
                   const float* __restrict__ lnw, const float* __restrict__ lnb,
                   unsigned short* __restrict__ out) {
  int gw = (blockIdx.x * blockDim.x + threadIdx.x) >> 5;
  int lane = threadIdx.x & 31;
  if (gw >= BLr * 4) return;
  int row = gw >> 2, h = gw & 3;
  size_t base = (size_t)row * Vd + h * DVs;
  float vals[8], s = 0.f, s2 = 0.f;
#pragma unroll
  for (int i = 0; i < 8; ++i) {
    float v = o[base + lane + i * 32];
    vals[i] = v; s += v; s2 += v * v;
  }
#pragma unroll
  for (int off = 16; off; off >>= 1) {
    s  += __shfl_xor(s, off, 32);
    s2 += __shfl_xor(s2, off, 32);
  }
  float mean = s * (1.f / DVs);
  float var  = s2 * (1.f / DVs) - mean * mean;
  float rstd = rsqrtf(var + 1e-5f);
#pragma unroll
  for (int i = 0; i < 8; ++i) {
    int dv = lane + i * 32;
    float y  = (vals[i] - mean) * rstd * lnw[dv] + lnb[dv];
    float gg = g[base + dv];
    y *= gg / (1.f + __expf(-gg));              // swish gate
    out[base + dv] = f2bf(y);
  }
}

// ---------------------------------------------------------------------------
// host launcher
// ---------------------------------------------------------------------------
extern "C" void kernel_launch(void* const* d_in, const int* in_sizes, int n_in,
                              void* d_out, int out_size, void* d_ws,
                              size_t ws_size, hipStream_t stream) {
  (void)in_sizes; (void)n_in; (void)out_size; (void)ws_size;
  const float* x    = (const float*)d_in[0];
  const float* mu_x = (const float*)d_in[1];
  const float* Wx1  = (const float*)d_in[2];
  const float* Wx2  = (const float*)d_in[3];
  const float* xb   = (const float*)d_in[4];
  const float* Wr   = (const float*)d_in[5];
  const float* Wk   = (const float*)d_in[6];
  const float* Wv   = (const float*)d_in[7];
  const float* Wg   = (const float*)d_in[8];
  const float* Ww1  = (const float*)d_in[9];
  const float* Ww2  = (const float*)d_in[10];
  const float* bw2  = (const float*)d_in[11];
  const float* bon  = (const float*)d_in[12];
  const float* lnw  = (const float*)d_in[13];
  const float* lnb  = (const float*)d_in[14];
  const float* Wo   = (const float*)d_in[15];

  char* p = (char*)d_ws;
  auto carve = [&](size_t bytes) -> void* {
    void* r = p; p += (bytes + 255) & ~(size_t)255; return r;
  };
  unsigned short* Wx1b = (unsigned short*)carve(160 * 1024 * 2);
  unsigned short* Wx2b = (unsigned short*)carve(1024 * 160 * 2);
  unsigned short* Wrb  = (unsigned short*)carve(512 * 1024 * 2);
  unsigned short* Wkb  = (unsigned short*)carve(512 * 1024 * 2);
  unsigned short* Wvb  = (unsigned short*)carve(1024 * 1024 * 2);
  unsigned short* Wgb  = (unsigned short*)carve(1024 * 1024 * 2);
  unsigned short* Ww1b = (unsigned short*)carve(64 * 1024 * 2);
  unsigned short* Ww2b = (unsigned short*)carve(512 * 64 * 2);
  unsigned short* Wob  = (unsigned short*)carve(1024 * 1024 * 2);
  float*          dlt  = (float*)carve((size_t)BLr * Hd * 4);
  unsigned short* t0b  = (unsigned short*)carve((size_t)BLr * Hd * 2);
  unsigned short* xpb  = (unsigned short*)carve((size_t)BLr * 160 * 2);
  unsigned short* xmix[5];
  for (int i = 0; i < 5; ++i) xmix[i] = (unsigned short*)carve((size_t)BLr * Hd * 2);
  float* rbuf = (float*)carve((size_t)BLr * Kd * 4);
  float* kbuf = (float*)carve((size_t)BLr * Kd * 4);
  float* wbuf = (float*)carve((size_t)BLr * Kd * 4);
  float* vbuf = (float*)carve((size_t)BLr * Vd * 4);
  float* gbuf = (float*)carve((size_t)BLr * Vd * 4);
  unsigned short* wtmp = (unsigned short*)carve((size_t)BLr * 64 * 2);
  float* obuf = (float*)carve((size_t)BLr * Vd * 4);
  unsigned short* gated = (unsigned short*)carve((size_t)BLr * Vd * 2);

  auto conv = [&](const float* s, unsigned short* d, int n) {
    f32_to_bf16_kernel<<<(n + 255) / 256, 256, 0, stream>>>(s, d, n);
  };
  conv(Wx1, Wx1b, 160 * 1024);
  conv(Wx2, Wx2b, 1024 * 160);
  conv(Wr, Wrb, 512 * 1024);
  conv(Wk, Wkb, 512 * 1024);
  conv(Wv, Wvb, 1024 * 1024);
  conv(Wg, Wgb, 1024 * 1024);
  conv(Ww1, Ww1b, 64 * 1024);
  conv(Ww2, Ww2b, 512 * 64);
  conv(Wo, Wob, 1024 * 1024);

  prep_kernel<<<(BLr * Hd + 255) / 256, 256, 0, stream>>>(x, mu_x, dlt, t0b);

  // xp = tanh(t0 @ Wx1^T)  [4096,160]
  gemm_wmma<1, unsigned short><<<dim3(BLr / 128, (160 + 63) / 64), 256, 0, stream>>>(
      t0b, Hd, Wx1b, Hd, xpb, 160, BLr, 160, Hd, nullptr, nullptr, nullptr);

  // xmix_n = bf16(x + delta*(xp_n @ Wx2_n^T + x_bias[n]))
  for (int n = 0; n < 5; ++n) {
    gemm_wmma<2, unsigned short><<<dim3(BLr / 128, Hd / 64), 256, 0, stream>>>(
        xpb + n * 32, 160, Wx2b + n * 32, 160, xmix[n], Hd, BLr, Hd, 32,
        xb + n * Hd, x, dlt);
  }

  // r/k/v/g projections
  gemm_wmma<0, float><<<dim3(BLr / 128, Kd / 64), 256, 0, stream>>>(
      xmix[0], Hd, Wrb, Hd, rbuf, Kd, BLr, Kd, Hd, nullptr, nullptr, nullptr);
  gemm_wmma<0, float><<<dim3(BLr / 128, Kd / 64), 256, 0, stream>>>(
      xmix[2], Hd, Wkb, Hd, kbuf, Kd, BLr, Kd, Hd, nullptr, nullptr, nullptr);
  gemm_wmma<0, float><<<dim3(BLr / 128, Vd / 64), 256, 0, stream>>>(
      xmix[3], Hd, Wvb, Hd, vbuf, Vd, BLr, Vd, Hd, nullptr, nullptr, nullptr);
  gemm_wmma<0, float><<<dim3(BLr / 128, Vd / 64), 256, 0, stream>>>(
      xmix[4], Hd, Wgb, Hd, gbuf, Vd, BLr, Vd, Hd, nullptr, nullptr, nullptr);

  // w LoRA: wtmp = tanh(xmix_w @ Ww1^T); w = wtmp @ Ww2^T + bw2
  gemm_wmma<1, unsigned short><<<dim3(BLr / 128, 1), 256, 0, stream>>>(
      xmix[1], Hd, Ww1b, Hd, wtmp, 64, BLr, 64, Hd, nullptr, nullptr, nullptr);
  gemm_wmma<0, float><<<dim3(BLr / 128, Kd / 64), 256, 0, stream>>>(
      wtmp, 64, Ww2b, 64, wbuf, Kd, BLr, Kd, 64, bw2, nullptr, nullptr);

  // chunked recurrent scan (fp32 WMMA, state in LDS)
  scan_kernel<<<dim3(8), dim3(256), SCAN_SMEM_FLOATS * 4, stream>>>(
      rbuf, kbuf, wbuf, vbuf, bon, obuf);

  // LN + swish gate -> bf16
  lngate_kernel<<<(BLr * 4 * 32 + 255) / 256, 256, 0, stream>>>(
      obuf, gbuf, lnw, lnb, gated);

  // out = gated @ Wo^T -> d_out (fp32)
  gemm_wmma<0, float><<<dim3(BLr / 128, Hd / 64), 256, 0, stream>>>(
      gated, Vd, Wob, Vd, (float*)d_out, Hd, BLr, Hd, Vd, nullptr, nullptr,
      nullptr);
}
